// GPTAttention_15109694948110
// MI455X (gfx1250) — matrix operand order repaired and checked by reading, not compiled, not run on hardware
//
#include <hip/hip_runtime.h>

// ---------------- problem constants (from reference) ----------------
#define BB 2
#define SS 2048
#define EE 1024
#define HH 16
#define DD 64   // EE/HH

typedef unsigned int   u32;
typedef unsigned short u16;

typedef __bf16 bf16_t;
typedef bf16_t v16bf __attribute__((ext_vector_type(16)));
typedef float  v8f   __attribute__((ext_vector_type(8)));
typedef u32    v8u   __attribute__((ext_vector_type(8)));
typedef u32    v4u   __attribute__((ext_vector_type(4)));
typedef u16    v16h  __attribute__((ext_vector_type(16)));

// ---------------- gfx1250 async global->LDS copy (guarded) ----------------
#if defined(__gfx1250__) && __has_builtin(__builtin_amdgcn_global_load_async_to_lds_b128)
#define HAVE_ASYNC_LDS 1
typedef int v4i __attribute__((vector_size(16)));
__device__ __forceinline__ void async_cp16(const void* g, void* l) {
    __builtin_amdgcn_global_load_async_to_lds_b128(
        (__attribute__((address_space(1))) v4i*)g,
        (__attribute__((address_space(3))) v4i*)l, 0, 0);
}
#if __has_builtin(__builtin_amdgcn_s_wait_asynccnt)
#define WAIT_ASYNC() __builtin_amdgcn_s_wait_asynccnt(0)
#else
#define WAIT_ASYNC() asm volatile("s_wait_asynccnt 0" ::: "memory")
#endif
#else
#define HAVE_ASYNC_LDS 0
#endif

// ---------------- helpers ----------------
__device__ __forceinline__ u16 f2bf(float f) {
    u32 u = __float_as_uint(f);
    u += 0x7fffu + ((u >> 16) & 1u);   // round-to-nearest-even
    return (u16)(u >> 16);
}

// 16-bit A-matrix 16x32 pair base per VGPR index p (K offset of even element,
// before adding 8*khalf): VGPR0..3 -> K 0,2,4,6 ; VGPR4..7 -> K 16,18,20,22
__device__ __forceinline__ int kpair_base(int p) {
    return (p < 4) ? (2 * p) : (16 + 2 * (p - 4));
}

__device__ __forceinline__ float rowred_max16(float v) {
    #pragma unroll
    for (int off = 1; off < 16; off <<= 1)
        v = fmaxf(v, __shfl_xor(v, off, 32));
    return v;
}
__device__ __forceinline__ float rowred_sum16(float v) {
    #pragma unroll
    for (int off = 1; off < 16; off <<= 1)
        v += __shfl_xor(v, off, 32);
    return v;
}

// ---------------- conversion kernels ----------------
__global__ __launch_bounds__(256) void cvt_f32_bf16(const float* __restrict__ in,
                                                    u16* __restrict__ out, int n) {
    for (int i = blockIdx.x * blockDim.x + threadIdx.x; i < n; i += gridDim.x * blockDim.x)
        out[i] = f2bf(in[i]);
}

// in: f32 [K,N] row-major  ->  out: bf16 [N,K] row-major (i.e. B^T)
__global__ __launch_bounds__(256) void cvt_transpose_f32_bf16(const float* __restrict__ in,
                                                              u16* __restrict__ out,
                                                              int K, int N) {
    int total = K * N;
    for (int i = blockIdx.x * blockDim.x + threadIdx.x; i < total; i += gridDim.x * blockDim.x) {
        int n = i % N;
        int k = i / N;
        out[(size_t)n * K + k] = f2bf(in[i]);   // coalesced read along n
    }
}

// ---------------- WMMA GEMM: C[M,N] = A[M,K] * B[K,N] + bias ----------------
// A: bf16 [M,K] row-major.  Bt: bf16 [N,K] row-major (pre-transposed B).
// Block = 256 threads (8 waves) computing a 128x64 tile; per K=32 step the
// A(128x32) and Bt(64x32) tiles are staged in LDS (double buffered, async
// global->LDS copies when available) and shared by all 8 waves.
#define LDSTRIDE 40   // bf16 elements: 80B row stride (16B aligned, bank-spread)

template <bool OUT_BF16>
__global__ __launch_bounds__(256) void gemm_wmma_bf16(const u16* __restrict__ A,
                                                      const u16* __restrict__ Bt,
                                                      const float* __restrict__ bias,
                                                      void* __restrict__ Cout,
                                                      int M, int N, int K) {
    __shared__ u16 Alds[2][128 * LDSTRIDE];
    __shared__ u16 Blds[2][64 * LDSTRIDE];

    const int tid   = threadIdx.x;
    const int lane  = tid & 31;
    const int wave  = tid >> 5;
    const int nIdx  = lane & 15;
    const int khalf = lane >> 4;
    const int m0 = blockIdx.y * 128;
    const int n0 = blockIdx.x * 64;

    // staging coordinates: A = 512 16B chunks (2/thread), B = 256 (1/thread)
    const int aRow0 = (tid * 2) >> 2, aCol0 = (tid * 2) & 3;       // chunk j=0
    const int aRow1 = (tid * 2 + 1) >> 2, aCol1 = (tid * 2 + 1) & 3;
    const int bRow  = tid >> 2, bCol = tid & 3;

    auto stage = [&](int buf, int k0) {
        const u16* gA0 = A  + (size_t)(m0 + aRow0) * K + k0 + aCol0 * 8;
        const u16* gA1 = A  + (size_t)(m0 + aRow1) * K + k0 + aCol1 * 8;
        const u16* gB  = Bt + (size_t)(n0 + bRow)  * K + k0 + bCol * 8;
        u16* lA0 = &Alds[buf][aRow0 * LDSTRIDE + aCol0 * 8];
        u16* lA1 = &Alds[buf][aRow1 * LDSTRIDE + aCol1 * 8];
        u16* lB  = &Blds[buf][bRow * LDSTRIDE + bCol * 8];
#if HAVE_ASYNC_LDS
        async_cp16(gA0, lA0);
        async_cp16(gA1, lA1);
        async_cp16(gB,  lB);
#else
        v4u a0 = *(const v4u*)gA0;
        v4u a1 = *(const v4u*)gA1;
        v4u b0 = *(const v4u*)gB;
        *(v4u*)lA0 = a0;
        *(v4u*)lA1 = a1;
        *(v4u*)lB  = b0;
#endif
        if (k0 + 64 < K) {   // prefetch one tile further ahead
            __builtin_prefetch(gA0 + 32, 0, 1);
            __builtin_prefetch(gB + 32, 0, 1);
        }
    };

    v8f acc[4];
    #pragma unroll
    for (int t = 0; t < 4; ++t)
        #pragma unroll
        for (int r = 0; r < 8; ++r) acc[t][r] = 0.0f;

    stage(0, 0);
    int cur = 0;

    for (int k0 = 0; k0 < K; k0 += 32) {
#if HAVE_ASYNC_LDS
        WAIT_ASYNC();                 // own copies landed in LDS
#endif
        __syncthreads();              // everyone's copies visible
        if (k0 + 32 < K) stage(cur ^ 1, k0 + 32);

        const u32* Al = (const u32*)&Alds[cur][0];
        const u32* Bl = (const u32*)&Blds[cur][0];

        // A fragment: lane holds row (wave*16 + lane&15), K pairs per layout
        v8u araw;
        #pragma unroll
        for (int p = 0; p < 8; ++p) {
            int kk = kpair_base(p) + 8 * khalf;
            araw[p] = Al[(wave * 16 + nIdx) * (LDSTRIDE / 2) + (kk >> 1)];
        }
        v16bf aF = __builtin_bit_cast(v16bf, araw);

        #pragma unroll
        for (int t = 0; t < 4; ++t) {
            // B fragment: lane holds col n, K pairs per layout
            v8u braw;
            #pragma unroll
            for (int p = 0; p < 8; ++p)
                braw[p] = Bl[(t * 16 + nIdx) * (LDSTRIDE / 2) + ((kpair_base(p) + 8 * khalf) >> 1)];
            v16bf bF = __builtin_bit_cast(v16bf, braw);
            acc[t] = __builtin_amdgcn_wmma_f32_16x16x32_bf16(
                false, aF, false, bF, (short)0, acc[t], false, false);
        }
        cur ^= 1;
    }

    // epilogue: C-layout -> lane holds col nIdx, rows r + 8*khalf
    #pragma unroll
    for (int t = 0; t < 4; ++t) {
        int n = n0 + t * 16 + nIdx;
        float bv = bias[n];
        #pragma unroll
        for (int r = 0; r < 8; ++r) {
            int m = m0 + wave * 16 + r + 8 * khalf;
            float v = acc[t][r] + bv;
            if (OUT_BF16) ((u16*)Cout)[(size_t)m * N + n] = f2bf(v);
            else          ((float*)Cout)[(size_t)m * N + n] = v;
        }
    }
}

// ---------------- flash attention (causal), one wave per 16-query tile ------
// qkv: bf16 [B*S, 3E] (Q at +0, K at +E, V at +2E). out: bf16 [B*S, E].
__global__ __launch_bounds__(128) void flash_attn_wmma(const u16* __restrict__ qkv,
                                                       u16* __restrict__ out) {
    __shared__ u16 Plds[4][16 * 33];     // per-wave P tile, padded stride 33

    const int lane  = threadIdx.x & 31;
    const int wave  = threadIdx.x >> 5;
    const int nIdx  = lane & 15;
    const int khalf = lane >> 4;

    const int task = blockIdx.x * 4 + wave;      // B*H*(S/16) tasks
    const int qt = task & 127;                   // S/16 = 128
    const int h  = (task >> 7) & 15;
    const int b  = task >> 11;
    const int q0 = qt * 16;
    const int tokBase = b * SS;
    const int ldq = 3 * EE;
    const u32* q32 = (const u32*)qkv;

    // Q fragments: A 16x64 = two 16x32 chunks; pairs contiguous along d
    v16bf aQ[2];
    #pragma unroll
    for (int c = 0; c < 2; ++c) {
        v8u raw;
        int tok = tokBase + q0 + nIdx;
        #pragma unroll
        for (int p = 0; p < 8; ++p) {
            int d = 32 * c + kpair_base(p) + 8 * khalf;
            raw[p] = q32[((size_t)tok * ldq + h * 64 + d) >> 1];
        }
        aQ[c] = __builtin_bit_cast(v16bf, raw);
    }

    v8f acc[4];
    #pragma unroll
    for (int t = 0; t < 4; ++t)
        #pragma unroll
        for (int r = 0; r < 8; ++r) acc[t][r] = 0.0f;
    float row_m[8], row_l[8];
    #pragma unroll
    for (int r = 0; r < 8; ++r) { row_m[r] = -INFINITY; row_l[r] = 0.0f; }

    for (int key0 = 0; key0 <= q0 + 15; key0 += 32) {
        // ---- scores: two 16x16 tiles covering keys key0..key0+31 ----
        v8f s[2];
        #pragma unroll
        for (int kt = 0; kt < 2; ++kt) {
            #pragma unroll
            for (int r = 0; r < 8; ++r) s[kt][r] = 0.0f;
            #pragma unroll
            for (int c = 0; c < 2; ++c) {
                // B = K^T (d x key): lane holds key col, d pairs per layout
                int key = key0 + kt * 16 + nIdx;
                key = key < (SS - 1) ? key : (SS - 1);
                v8u raw;
                #pragma unroll
                for (int p = 0; p < 8; ++p) {
                    int d = 32 * c + kpair_base(p) + 8 * khalf;
                    raw[p] = q32[((size_t)(tokBase + key) * ldq + EE + h * 64 + d) >> 1];
                }
                v16bf bK = __builtin_bit_cast(v16bf, raw);
                s[kt] = __builtin_amdgcn_wmma_f32_16x16x32_bf16(
                    false, aQ[c], false, bK, (short)0, s[kt], false, false);
            }
        }

        // ---- online softmax (rows live in half-wave groups of 16 lanes) ----
        #pragma unroll
        for (int r = 0; r < 8; ++r) {
            int m = q0 + r + 8 * khalf;
            float v0 = s[0][r] * 0.125f;                 // 1/sqrt(64)
            float v1 = s[1][r] * 0.125f;
            if (key0 + nIdx      > m) v0 = -INFINITY;    // causal mask
            if (key0 + 16 + nIdx > m) v1 = -INFINITY;
            float tmax = rowred_max16(fmaxf(v0, v1));
            float mnew = fmaxf(row_m[r], tmax);
            float alpha = __expf(row_m[r] - mnew);
            float p0 = __expf(v0 - mnew);
            float p1 = __expf(v1 - mnew);
            float rs = rowred_sum16(p0 + p1);
            row_l[r] = row_l[r] * alpha + rs;
            row_m[r] = mnew;
            #pragma unroll
            for (int t = 0; t < 4; ++t) acc[t][r] *= alpha;
            int prow = r + 8 * khalf;
            Plds[wave][prow * 33 + nIdx]      = f2bf(p0);
            Plds[wave][prow * 33 + 16 + nIdx] = f2bf(p1);
        }
        __builtin_amdgcn_wave_barrier();    // keep DS store->load order

        // ---- P (16x32) C-layout -> A-fragment via LDS ----
        v16h praw;
        #pragma unroll
        for (int p = 0; p < 8; ++p) {
            int kk = kpair_base(p) + 8 * khalf;
            praw[2 * p]     = Plds[wave][nIdx * 33 + kk];
            praw[2 * p + 1] = Plds[wave][nIdx * 33 + kk + 1];
        }
        v16bf aP = __builtin_bit_cast(v16bf, praw);
        __builtin_amdgcn_wave_barrier();

        // ---- O += P * V ; V is B (32 keys x 64 d), key pairs strided ----
        #pragma unroll
        for (int t = 0; t < 4; ++t) {
            int col = 2 * EE + h * 64 + t * 16 + nIdx;
            v16h vraw;
            #pragma unroll
            for (int p = 0; p < 8; ++p) {
                int k1 = key0 + kpair_base(p) + 8 * khalf;
                int ka = k1     < (SS - 1) ? k1     : (SS - 1);
                int kb = k1 + 1 < (SS - 1) ? k1 + 1 : (SS - 1);
                vraw[2 * p]     = qkv[(size_t)(tokBase + ka) * ldq + col];
                vraw[2 * p + 1] = qkv[(size_t)(tokBase + kb) * ldq + col];
            }
            v16bf bV = __builtin_bit_cast(v16bf, vraw);
            acc[t] = __builtin_amdgcn_wmma_f32_16x16x32_bf16(
                false, aP, false, bV, (short)0, acc[t], false, false);
        }
    }

    // ---- epilogue: normalize and store merged-head bf16 output ----
    #pragma unroll
    for (int r = 0; r < 8; ++r) {
        float inv = 1.0f / row_l[r];
        int tok = tokBase + q0 + r + 8 * khalf;
        #pragma unroll
        for (int t = 0; t < 4; ++t)
            out[(size_t)tok * EE + h * 64 + t * 16 + nIdx] = f2bf(acc[t][r] * inv);
    }
}

// ---------------- host-side launch ----------------
extern "C" void kernel_launch(void* const* d_in, const int* in_sizes, int n_in,
                              void* d_out, int out_size, void* d_ws, size_t ws_size,
                              hipStream_t stream) {
    (void)in_sizes; (void)n_in; (void)out_size; (void)ws_size;

    const float* hidden = (const float*)d_in[0];  // [B,S,E]
    const float* w_attn = (const float*)d_in[1];  // [E,3E]
    const float* b_attn = (const float*)d_in[2];  // [3E]
    const float* w_proj = (const float*)d_in[3];  // [E,E]
    const float* b_proj = (const float*)d_in[4];  // [E]
    float* outp = (float*)d_out;                  // [B,S,E] f32

    const int M  = BB * SS;       // 4096 tokens
    const int N3 = 3 * EE;        // 3072
    const int K  = EE;            // 1024

    u16* ws = (u16*)d_ws;
    size_t o = 0;
    u16* hsbf = ws + o; o += (size_t)M * EE;      // hidden bf16
    u16* watT = ws + o; o += (size_t)N3 * EE;     // w_attn^T bf16  [3E,E]
    u16* wpT  = ws + o; o += (size_t)EE * EE;     // w_proj^T bf16  [E,E]
    u16* qkv  = ws + o; o += (size_t)M * N3;      // qkv bf16       [M,3E]
    u16* aout = ws + o; o += (size_t)M * EE;      // attn out bf16  [M,E]

    cvt_f32_bf16<<<4096, 256, 0, stream>>>(hidden, hsbf, M * EE);
    cvt_transpose_f32_bf16<<<4096, 256, 0, stream>>>(w_attn, watT, EE, N3);
    cvt_transpose_f32_bf16<<<2048, 256, 0, stream>>>(w_proj, wpT, EE, EE);

    // QKV projection: [M,3E] = hs * w_attn + b_attn   (bf16 out)
    gemm_wmma_bf16<true><<<dim3(N3 / 64, M / 128), 256, 0, stream>>>(
        hsbf, watT, b_attn, (void*)qkv, M, N3, K);

    // causal flash attention, merged-head bf16 output
    flash_attn_wmma<<<(BB * HH * (SS / 16)) / 4, 128, 0, stream>>>(qkv, aout);

    // output projection: [M,E] = aout * w_proj + b_proj   (f32 out)
    gemm_wmma_bf16<false><<<dim3(EE / 64, M / 128), 256, 0, stream>>>(
        aout, wpT, b_proj, (void*)outp, M, EE, K);
}
